// Conv3dOptimized_34007551049783
// MI455X (gfx1250) — matrix operand order repaired and checked by reading, not compiled, not run on hardware
//
#include <hip/hip_runtime.h>

typedef __attribute__((ext_vector_type(16))) __bf16 v16bf;
typedef __attribute__((ext_vector_type(8)))  __bf16 v8bf;
typedef __attribute__((ext_vector_type(2)))  __bf16 v2bf;
typedef __attribute__((ext_vector_type(8)))  float  v8f;

#define H_     96
#define W_     96
#define D_     35
#define HWD    322560      // 96*96*35
#define CIN    32
#define COUT   64
#define XBSTR  10321920    // CIN*HWD
#define OBSTR  20643840    // COUT*HWD
#define NPOS   645120      // 2*HWD
#define NTOT   41287680    // 2*COUT*HWD
#define NTILE  64          // positions per block
#define WELEMS (27*COUT*CIN)   // 55296 weight elements

__device__ __forceinline__ unsigned short f2bf(float f) {
    __bf16 b = (__bf16)f;                       // hardware f32->bf16 (RNE)
    return __builtin_bit_cast(unsigned short, b);
}
__device__ __forceinline__ unsigned pack2bf(float a, float b) {
    v2bf t;
    t[0] = (__bf16)a;
    t[1] = (__bf16)b;
    return __builtin_bit_cast(unsigned, t);     // v_cvt_pk path
}

// ---------------------------------------------------------------------------
// Kernel 1: implicit-GEMM Conv3d via v_wmma_f32_16x16x32_bf16.
// Block = 256 threads (8 waves). Block tile: 64 co x 64 positions, K = 27*32.
// Full weight matrix staged once into LDS as bf16 [tap][o][ci] (110.6 KB).
// B tile double-buffered (one barrier per K-step). Tap loop fully unrolled.
// ---------------------------------------------------------------------------
__global__ void __launch_bounds__(256)
conv3d_wmma(const float* __restrict__ x, const float* __restrict__ w,
            const float* __restrict__ bias, float* __restrict__ out) {
    __shared__ __align__(16) unsigned short Wall[27 * COUT * CIN]; // [tap][o][ci]
    __shared__ __align__(32) unsigned short Bws[2][NTILE * CIN];   // [buf][pos][ci]

    const int tid  = threadIdx.x;
    const int lane = tid & 31;
    const int wv   = tid >> 5;
    const int mt   = wv & 3;              // M tile: co base = mt*16
    const int nt0  = (wv >> 2) * 2;       // N sub-tiles nt0, nt0+1 (16 pos each)
    const int tile_base = blockIdx.x * NTILE;

    // ---- stage ALL weights into LDS once (coalesced global reads) ----
    for (int e = tid; e < WELEMS; e += 256) {
        const int o   = e / 864;
        const int r   = e - o * 864;
        const int ci  = r / 27;
        const int tap = r - ci * 27;
        Wall[tap * (COUT * CIN) + o * CIN + ci] = f2bf(w[e]);
    }

    // --- per-thread decode for the B (activation) stage ---
    const int bp  = tid >> 2;                 // local position 0..63
    const int cij = (tid & 3) * 8;            // ci base, 8 per thread
    int s    = tile_base + bp;                // global position (always < NPOS)
    int bidx = s / HWD;
    int r0   = s - bidx * HWD;
    int h    = r0 / (W_ * D_);
    int r2   = r0 - h * (W_ * D_);
    int wc   = r2 / D_;
    int d    = r2 - wc * D_;
    const int base0 = bidx * XBSTR + h * (W_ * D_) + wc * D_ + d;

    // --- fragment-read constants ---
    const int oa = mt * 16 + (lane & 15);
    const int cb = (lane < 16) ? 0 : 8;
    const int kb = (lane < 16) ? 0 : 16;
    const int np0 = (nt0 * 16)      + (lane & 15);
    const int np1 = (nt0 * 16 + 16) + (lane & 15);

    v8f acc0 = {};
    v8f acc1 = {};

    #pragma unroll
    for (int tap = 0; tap < 27; ++tap) {
        const int dh = tap / 9 - 1;                    // compile-time constants
        const int rr = tap - (tap / 9) * 9;
        const int dw = rr / 3 - 1;
        const int dd = rr - (rr / 3) * 3 - 1;
        const int off = dh * (W_ * D_) + dw * D_ + dd; // immediate per tap
        const int buf = tap & 1;

        // ---- stage B tile: unconditional clause'd loads, zeroed at pack ----
        {
            const int hh = h + dh, ww = wc + dw, dv = d + dd;
            const bool ok = ((unsigned)hh < (unsigned)H_) &
                            ((unsigned)ww < (unsigned)W_) &
                            ((unsigned)dv < (unsigned)D_);
            const int aoff = ok ? (base0 + off) : 0;   // safe addr when OOB
            float v[8];
            #pragma unroll
            for (int j = 0; j < 8; ++j)
                v[j] = x[aoff + (cij + j) * HWD];
            const unsigned u0 = ok ? pack2bf(v[0], v[1]) : 0u;
            const unsigned u1 = ok ? pack2bf(v[2], v[3]) : 0u;
            const unsigned u2 = ok ? pack2bf(v[4], v[5]) : 0u;
            const unsigned u3 = ok ? pack2bf(v[6], v[7]) : 0u;
            *(uint4*)&Bws[buf][bp * CIN + cij] = make_uint4(u0, u1, u2, u3);
        }

        __syncthreads();   // single barrier per K-step (double-buffered B)

        // ---- A fragment (shared by both WMMAs of this wave) ----
        // A 16x32: lanes 0-15 rows M, K = 0..15; lanes 16-31 K = 16..31
        const unsigned short* wrow = &Wall[tap * (COUT * CIN) + oa * CIN];
        union { v16bf v; v8bf half[2]; } afr;
        afr.half[0] = *(const v8bf*)&wrow[kb];
        afr.half[1] = *(const v8bf*)&wrow[kb + 8];

        // ---- B fragments: lanes 0-15 K=0..15, lanes 16-31 K=16..31, N = lane%16 ----
        v16bf bfr0 = *(const v16bf*)&Bws[buf][np0 * CIN + kb];
        v16bf bfr1 = *(const v16bf*)&Bws[buf][np1 * CIN + kb];

        acc0 = __builtin_amdgcn_wmma_f32_16x16x32_bf16(
                   false, afr.v, false, bfr0, (short)0, acc0, false, false);
        acc1 = __builtin_amdgcn_wmma_f32_16x16x32_bf16(
                   false, afr.v, false, bfr1, (short)0, acc1, false, false);
    }

    // ---- epilogue: C/D layout VGPR r, lane l -> M = r + (l<16?0:8), N = l%16 ----
    const int co0 = mt * 16 + ((lane < 16) ? 0 : 8);
    #pragma unroll
    for (int u = 0; u < 2; ++u) {
        const int sPos = tile_base + (nt0 + u) * 16 + (lane & 15);
        const int bo   = sPos / HWD;
        const int sp   = sPos - bo * HWD;
        const int obase = bo * OBSTR + sp;
        const v8f acc = u ? acc1 : acc0;
        #pragma unroll
        for (int rr = 0; rr < 8; ++rr) {
            const int co = co0 + rr;
            out[obase + co * HWD] = acc[rr] + bias[co];
        }
    }
}

// ---------------------------------------------------------------------------
// Kernel 2: deterministic per-channel mean / inv_std (one block per channel).
// ---------------------------------------------------------------------------
__global__ void __launch_bounds__(256)
bn_stats(const float* __restrict__ out, float* __restrict__ st) {
    __shared__ float ssum[256];
    __shared__ float ssq[256];
    const int c = blockIdx.x;
    float s = 0.f, q = 0.f;
    for (int b = 0; b < 2; ++b) {
        const float* p = out + b * OBSTR + c * HWD;
        for (int i = threadIdx.x; i < HWD; i += 256) {
            float v = p[i];
            s += v;
            q += v * v;
        }
    }
    ssum[threadIdx.x] = s;
    ssq[threadIdx.x]  = q;
    __syncthreads();
    for (int stp = 128; stp > 0; stp >>= 1) {
        if (threadIdx.x < stp) {
            ssum[threadIdx.x] += ssum[threadIdx.x + stp];
            ssq[threadIdx.x]  += ssq[threadIdx.x + stp];
        }
        __syncthreads();
    }
    if (threadIdx.x == 0) {
        const float n    = (float)NPOS;
        const float mean = ssum[0] / n;
        const float var  = ssq[0] / n - mean * mean;
        st[c]        = mean;
        st[COUT + c] = rsqrtf(var + 1e-5f);
    }
}

// ---------------------------------------------------------------------------
// Kernel 3: in-place BatchNorm affine + LeakyReLU(0.2), float4 vectorized.
// ---------------------------------------------------------------------------
__global__ void __launch_bounds__(256)
bn_lrelu(float* __restrict__ out, const float* __restrict__ st,
         const float* __restrict__ gamma, const float* __restrict__ beta) {
    const int i4 = blockIdx.x * 256 + threadIdx.x;   // float4 index
    const int i  = i4 * 4;                            // HWD % 4 == 0 -> one channel per vec
    const int co = (i / HWD) & (COUT - 1);
    const float g  = gamma[co] * st[COUT + co];
    const float bb = beta[co] - st[co] * g;
    float4 v = ((const float4*)out)[i4];
    float a0 = v.x * g + bb, a1 = v.y * g + bb, a2 = v.z * g + bb, a3 = v.w * g + bb;
    v.x = a0 >= 0.f ? a0 : 0.2f * a0;
    v.y = a1 >= 0.f ? a1 : 0.2f * a1;
    v.z = a2 >= 0.f ? a2 : 0.2f * a2;
    v.w = a3 >= 0.f ? a3 : 0.2f * a3;
    ((float4*)out)[i4] = v;
}

// ---------------------------------------------------------------------------
extern "C" void kernel_launch(void* const* d_in, const int* in_sizes, int n_in,
                              void* d_out, int out_size, void* d_ws, size_t ws_size,
                              hipStream_t stream) {
    const float* x     = (const float*)d_in[0];
    const float* w     = (const float*)d_in[1];
    const float* b     = (const float*)d_in[2];
    const float* gamma = (const float*)d_in[3];
    const float* beta  = (const float*)d_in[4];
    float* out = (float*)d_out;
    float* st  = (float*)d_ws;     // 128 floats: [mean x64 | inv_std x64]

    conv3d_wmma<<<NPOS / NTILE, 256, 0, stream>>>(x, w, b, out);
    bn_stats  <<<COUT,          256, 0, stream>>>(out, st);
    bn_lrelu  <<<NTOT / 4 / 256, 256, 0, stream>>>(out, st, gamma, beta);
}
